// HyPEShortConvBlock_34110630265261
// MI455X (gfx1250) — matrix operand order repaired and checked by reading, compiled, not run on hardware
//
#include <hip/hip_runtime.h>
#include <hip/hip_bf16.h>

typedef __attribute__((ext_vector_type(16))) __bf16 v16bf;
typedef __attribute__((ext_vector_type(8)))  float  v8f;

#define D_MODEL 1024
#define D_CONV 1024
#define FFN_INNER 2816
#define CONV_K 3
#define HEAD_DIM 64
#define PAIRS 32
#define T_SEQ 4096
#define M_TOK 16384   // B*T

// Async global->LDS (gfx1250 GLOBAL_LOAD_ASYNC_TO_LDS_B128, ASYNCcnt) —
// enabled only if the device toolchain declares the builtins.
#if defined(__gfx1250__) && \
    __has_builtin(__builtin_amdgcn_global_load_async_to_lds_b128) && \
    __has_builtin(__builtin_amdgcn_s_wait_asynccnt)
#define HAVE_ASYNC_LDS 1
// b128 payload type per hipcc diagnostic: GCC-vector int4, AS(1) src / AS(3) dst
typedef int v4i_vs __attribute__((vector_size(16)));
typedef __attribute__((address_space(1))) v4i_vs* async_gptr;
typedef __attribute__((address_space(3))) v4i_vs* async_lptr;
#else
#define HAVE_ASYNC_LDS 0
#endif

// ---------------------------------------------------------------------------
// fp32 [R][C] -> bf16 [C][R]   (weight transpose to B = [K][N] row-major)
// ---------------------------------------------------------------------------
__global__ void k_transpose_bf16(const float* __restrict__ w, __bf16* __restrict__ wt,
                                 int R, int C) {
    int g = blockIdx.x * blockDim.x + threadIdx.x;
    int total = R * C;
    if (g >= total) return;
    int r = g / C;
    int c = g - r * C;
    wt[(size_t)c * R + r] = (__bf16)w[(size_t)r * C + c];
}

// ---------------------------------------------------------------------------
// RMSNorm: fp32 [row][1024] * w -> bf16.  One block (256 thr) per row.
// ---------------------------------------------------------------------------
__global__ __launch_bounds__(256)
void k_rmsnorm(const float* __restrict__ in, const float* __restrict__ w,
               __bf16* __restrict__ out) {
    const int row = blockIdx.x;
    const float* xr = in + (size_t)row * D_MODEL;
    float vals[4];
    float acc = 0.f;
#pragma unroll
    for (int i = 0; i < 4; ++i) {
        vals[i] = xr[threadIdx.x + 256 * i];
        acc += vals[i] * vals[i];
    }
    __shared__ float red[256];
    red[threadIdx.x] = acc;
    __syncthreads();
    for (int s = 128; s > 0; s >>= 1) {
        if (threadIdx.x < s) red[threadIdx.x] += red[threadIdx.x + s];
        __syncthreads();
    }
    float rs = rsqrtf(red[0] * (1.0f / D_MODEL) + 1e-6f);
#pragma unroll
    for (int i = 0; i < 4; ++i) {
        int c = threadIdx.x + 256 * i;
        out[(size_t)row * D_MODEL + c] = (__bf16)(vals[i] * rs * w[c]);
    }
}

// ---------------------------------------------------------------------------
// Fused RoPE-gate + causal depthwise conv1d(K=3) + c*z.
// proj bf16 [M][3072]: cols [0,1024)=b, [1024,2048)=c, [2048,3072)=h.
// One thread per (token, pair) -> writes even/odd channels.
// ---------------------------------------------------------------------------
__global__ __launch_bounds__(256)
void k_rope_conv(const __bf16* __restrict__ proj,
                 const float* __restrict__ fcos, const float* __restrict__ fsin,
                 const float* __restrict__ convw, const float* __restrict__ convb,
                 __bf16* __restrict__ cz) {
    int g = blockIdx.x * blockDim.x + threadIdx.x;          // [0, M*512)
    int token = g >> 9;
    int pairIdx = g & 511;
    int head = pairIdx >> 5;
    int p = pairIdx & 31;
    int t = token & (T_SEQ - 1);                            // within-batch time
    int ch_e = head * HEAD_DIM + 2 * p;
    int ch_o = ch_e + 1;

    float ze = 0.f, zo = 0.f;
#pragma unroll
    for (int dt = 0; dt < CONV_K; ++dt) {
        int tau = t - (CONV_K - 1) + dt;
        if (tau < 0) continue;
        size_t tok2 = (size_t)(token - (CONV_K - 1) + dt);
        const __bf16* pr = proj + tok2 * 3072;
        float be = (float)pr[ch_e];
        float bo = (float)pr[ch_o];
        float he = (float)pr[2048 + ch_e];
        float ho = (float)pr[2048 + ch_o];
        float cc = fcos[(size_t)tau * PAIRS + p];
        float ss = fsin[(size_t)tau * PAIRS + p];
        float rot_e = be * cc - bo * ss;
        float rot_o = be * ss + bo * cc;
        ze += convw[ch_e * CONV_K + dt] * (rot_e * he);
        zo += convw[ch_o * CONV_K + dt] * (rot_o * ho);
    }
    ze += convb[ch_e];
    zo += convb[ch_o];
    const __bf16* prc = proj + (size_t)token * 3072 + 1024;
    float ce = (float)prc[ch_e];
    float co = (float)prc[ch_o];
    __bf16* dst = cz + (size_t)token * D_CONV;
    dst[ch_e] = (__bf16)(ce * ze);
    dst[ch_o] = (__bf16)(co * zo);
}

// ---------------------------------------------------------------------------
// WMMA GEMM: C[M,N] = A[M,K] x B[K,N], A/B bf16 row-major, f32 accumulate.
// Block: 256 thr (8 waves).  K-step 32.
// MODE 0: block 128x128, wave 32x64 (2x4 tiles), store bf16.
// MODE 1: block 128x128, wave 32x64, out f32 = aux + acc.
// MODE 2: block 128x64, wave 32x32 per B, dual-B, out bf16 = silu(g)*u.
// A tile fill uses GLOBAL_LOAD_ASYNC_TO_LDS_B128 when available (ASYNCcnt),
// B tile is stored in LDS pre-transposed: Bt[n][k] (k-stride padded to 40),
// so every fragment read (A and B) is a pair of ds_load_b128.
// ---------------------------------------------------------------------------
template<int MODE>
__global__ __launch_bounds__(256)
void k_gemm_wmma(const __bf16* __restrict__ A,
                 const __bf16* __restrict__ B0,
                 const __bf16* __restrict__ B1,
                 const float*  __restrict__ aux,
                 void* __restrict__ out,
                 int M, int N, int K) {
    constexpr int NSUB  = (MODE == 2) ? 2 : 4;     // 16-wide n-subtiles per wave
    constexpr int BN    = (MODE == 2) ? 64 : 128;  // block tile N
    constexpr int LDA_S = 40;                      // bf16 stride, A tile rows (128x32)
    constexpr int LDB_S = 40;                      // bf16 stride, Bt rows (k-dim 32 pad 40)
    __shared__ __bf16 As[128 * LDA_S];
    __shared__ __bf16 Bt0[BN * LDB_S];
    __shared__ __bf16 Bt1[(MODE == 2) ? BN * LDB_S : 1];

    const int tid = threadIdx.x;
    const int lane = tid & 31;
    const int lanelo = lane & 15;
    const int lanehi = lane >> 4;
    const int wv = tid >> 5;
    const int wm = wv & 3;
    const int wn = wv >> 2;
    const int m0 = blockIdx.y * 128;
    const int n0 = blockIdx.x * BN;

    v8f acc0[2][NSUB];
    v8f acc1[2][(MODE == 2) ? NSUB : 1];
#pragma unroll
    for (int i = 0; i < 2; ++i)
#pragma unroll
        for (int j = 0; j < NSUB; ++j)
#pragma unroll
            for (int r = 0; r < 8; ++r) {
                acc0[i][j][r] = 0.f;
                if (MODE == 2) acc1[i][j][r] = 0.f;
            }

    const int arow = tid >> 1, ahalf = tid & 1;    // A loader mapping

    for (int k0 = 0; k0 < K; k0 += 32) {
        __syncthreads();
        // ---- A tile: 32 bytes per thread, row-major in LDS ----
        {
            const __bf16* gsrc = A + (size_t)(m0 + arow) * K + k0 + ahalf * 16;
            __bf16* ldst = &As[arow * LDA_S + ahalf * 16];
#if HAVE_ASYNC_LDS
            __builtin_amdgcn_global_load_async_to_lds_b128(
                (async_gptr)gsrc, (async_lptr)ldst, 0, 0);
            __builtin_amdgcn_global_load_async_to_lds_b128(
                (async_gptr)(gsrc + 8), (async_lptr)(ldst + 8), 0, 0);
#else
            ((uint4*)ldst)[0] = ((const uint4*)gsrc)[0];
            ((uint4*)ldst)[1] = ((const uint4*)gsrc)[1];
#endif
            if (k0 + 32 < K)
                __builtin_prefetch(gsrc + 32, 0, 0);   // global_prefetch_b8
        }
        // ---- B tile: load 2(k) x 8(n) micro-tile, store transposed Bt[n][k]
        //      (k,k+1) for one n are contiguous -> 8 x ds_store_b32 ----
        if (MODE == 2) {
            const int tt = tid & 127;
            const __bf16* Bsel = (tid < 128) ? B0 : B1;
            __bf16* Bdst = (tid < 128) ? Bt0 : Bt1;
            const int prow = tt >> 3, seg = tt & 7;
            uint4 lo = *(const uint4*)(Bsel + (size_t)(k0 + 2 * prow) * N + n0 + seg * 8);
            uint4 hi = *(const uint4*)(Bsel + (size_t)(k0 + 2 * prow + 1) * N + n0 + seg * 8);
            const unsigned short* l = (const unsigned short*)&lo;
            const unsigned short* h = (const unsigned short*)&hi;
#pragma unroll
            for (int j = 0; j < 8; ++j) {
                unsigned v = (unsigned)l[j] | ((unsigned)h[j] << 16);
                *(unsigned*)&Bdst[(seg * 8 + j) * LDB_S + 2 * prow] = v;
            }
        } else {
            const int prow = tid >> 4, seg = tid & 15;
            uint4 lo = *(const uint4*)(B0 + (size_t)(k0 + 2 * prow) * N + n0 + seg * 8);
            uint4 hi = *(const uint4*)(B0 + (size_t)(k0 + 2 * prow + 1) * N + n0 + seg * 8);
            const unsigned short* l = (const unsigned short*)&lo;
            const unsigned short* h = (const unsigned short*)&hi;
#pragma unroll
            for (int j = 0; j < 8; ++j) {
                unsigned v = (unsigned)l[j] | ((unsigned)h[j] << 16);
                *(unsigned*)&Bt0[(seg * 8 + j) * LDB_S + 2 * prow] = v;
            }
        }
#if HAVE_ASYNC_LDS
        __builtin_amdgcn_s_wait_asynccnt(0);   // our async LDS writes landed
#endif
        __syncthreads();

        // ---- fragments: all contiguous b128 pairs ----
        union FR { v16bf v; __bf16 e[16]; uint4 q[2]; };
        FR af[2];
#pragma unroll
        for (int sm = 0; sm < 2; ++sm) {
            const __bf16* ap = &As[(wm * 32 + sm * 16 + lanelo) * LDA_S];
            const int koff = lanehi * 8;
            af[sm].q[0] = *(const uint4*)&ap[koff];       // K 0..7  / 8..15
            af[sm].q[1] = *(const uint4*)&ap[16 + koff];  // K 16..23 / 24..31
        }
        FR bf0[NSUB], bf1[NSUB];
#pragma unroll
        for (int sn = 0; sn < NSUB; ++sn) {
            const int col = wn * (NSUB * 16) + sn * 16 + lanelo;
            const __bf16* bp = &Bt0[col * LDB_S + lanehi * 8];
            bf0[sn].q[0] = *(const uint4*)&bp[0];
            bf0[sn].q[1] = *(const uint4*)&bp[16];
            if (MODE == 2) {
                const __bf16* bp1 = &Bt1[col * LDB_S + lanehi * 8];
                bf1[sn].q[0] = *(const uint4*)&bp1[0];
                bf1[sn].q[1] = *(const uint4*)&bp1[16];
            }
        }

        // ---- matrix cores ----
#pragma unroll
        for (int sm = 0; sm < 2; ++sm)
#pragma unroll
            for (int sn = 0; sn < NSUB; ++sn) {
                acc0[sm][sn] = __builtin_amdgcn_wmma_f32_16x16x32_bf16(
                    false, af[sm].v, false, bf0[sn].v,
                    (short)0, acc0[sm][sn], false, false);
                if (MODE == 2)
                    acc1[sm][sn] = __builtin_amdgcn_wmma_f32_16x16x32_bf16(
                        false, af[sm].v, false, bf1[sn].v,
                        (short)0, acc1[sm][sn], false, false);
            }
    }

    // ---- epilogue (C layout: VGPR r -> M = r + 8*lanehi, N = lanelo) ----
#pragma unroll
    for (int sm = 0; sm < 2; ++sm)
#pragma unroll
        for (int sn = 0; sn < NSUB; ++sn) {
            const int ncol = n0 + wn * (NSUB * 16) + sn * 16 + lanelo;
#pragma unroll
            for (int r = 0; r < 8; ++r) {
                const int mrow = m0 + wm * 32 + sm * 16 + lanehi * 8 + r;
                const size_t idx = (size_t)mrow * N + ncol;
                if (MODE == 0) {
                    ((__bf16*)out)[idx] = (__bf16)acc0[sm][sn][r];
                } else if (MODE == 1) {
                    ((float*)out)[idx] = aux[idx] + acc0[sm][sn][r];
                } else {
                    float g = acc0[sm][sn][r];
                    float u = acc1[sm][sn][r];
                    float s = g / (1.0f + __expf(-g));   // silu
                    ((__bf16*)out)[idx] = (__bf16)(s * u);
                }
            }
        }
}

// ---------------------------------------------------------------------------
extern "C" void kernel_launch(void* const* d_in, const int* in_sizes, int n_in,
                              void* d_out, int out_size, void* d_ws, size_t ws_size,
                              hipStream_t stream) {
    (void)in_sizes; (void)n_in; (void)out_size; (void)ws_size;
    const float* x     = (const float*)d_in[0];
    const float* fcos  = (const float*)d_in[1];
    const float* fsin  = (const float*)d_in[2];
    const float* wn1   = (const float*)d_in[3];
    const float* wproj = (const float*)d_in[4];
    const float* convw = (const float*)d_in[5];
    const float* convb = (const float*)d_in[6];
    const float* wout  = (const float*)d_in[7];
    const float* wn2   = (const float*)d_in[8];
    const float* wgate = (const float*)d_in[9];
    const float* wup   = (const float*)d_in[10];
    const float* wdown = (const float*)d_in[11];
    float* out = (float*)d_out;

    char* ws = (char*)d_ws;
    size_t off = 0;
    auto take = [&](size_t bytes) -> char* {
        char* p = ws + off;
        off += (bytes + 255) & ~(size_t)255;
        return p;
    };
    __bf16* wprojT = (__bf16*)take((size_t)D_MODEL * 3 * D_CONV * 2);
    __bf16* woutT  = (__bf16*)take((size_t)D_CONV * D_MODEL * 2);
    __bf16* wgateT = (__bf16*)take((size_t)D_MODEL * FFN_INNER * 2);
    __bf16* wupT   = (__bf16*)take((size_t)D_MODEL * FFN_INNER * 2);
    __bf16* wdownT = (__bf16*)take((size_t)FFN_INNER * D_MODEL * 2);
    __bf16* normed = (__bf16*)take((size_t)M_TOK * D_MODEL * 2);
    __bf16* projb  = (__bf16*)take((size_t)M_TOK * 3 * D_CONV * 2);
    __bf16* czb    = (__bf16*)take((size_t)M_TOK * D_CONV * 2);
    float*  resid  = (float*)take((size_t)M_TOK * D_MODEL * 4);
    __bf16* hbuf = normed;   // normed dead after proj GEMM
    __bf16* actb = projb;    // proj dead after rope_conv (act is smaller)

    // 1) weights -> bf16 B = [K][N]
    {
        int t1 = 3 * D_CONV * D_MODEL;
        k_transpose_bf16<<<(t1 + 255) / 256, 256, 0, stream>>>(wproj, wprojT, 3 * D_CONV, D_MODEL);
        int t2 = D_MODEL * D_CONV;
        k_transpose_bf16<<<(t2 + 255) / 256, 256, 0, stream>>>(wout, woutT, D_MODEL, D_CONV);
        int t3 = FFN_INNER * D_MODEL;
        k_transpose_bf16<<<(t3 + 255) / 256, 256, 0, stream>>>(wgate, wgateT, FFN_INNER, D_MODEL);
        k_transpose_bf16<<<(t3 + 255) / 256, 256, 0, stream>>>(wup, wupT, FFN_INNER, D_MODEL);
        int t4 = D_MODEL * FFN_INNER;
        k_transpose_bf16<<<(t4 + 255) / 256, 256, 0, stream>>>(wdown, wdownT, D_MODEL, FFN_INNER);
    }

    // 2) normed = rmsnorm(x, w_norm1)
    k_rmsnorm<<<M_TOK, 256, 0, stream>>>(x, wn1, normed);

    // 3) proj = normed @ w_proj^T    [16384 x 3072]
    k_gemm_wmma<0><<<dim3(3 * D_CONV / 128, M_TOK / 128), 256, 0, stream>>>(
        normed, wprojT, nullptr, nullptr, projb, M_TOK, 3 * D_CONV, D_MODEL);

    // 4) cz = c * causal_dwconv(rope_gate(b, h))
    k_rope_conv<<<(M_TOK * 512) / 256, 256, 0, stream>>>(projb, fcos, fsin, convw, convb, czb);

    // 5) resid = x + cz @ w_out^T
    k_gemm_wmma<1><<<dim3(D_MODEL / 128, M_TOK / 128), 256, 0, stream>>>(
        czb, woutT, nullptr, x, resid, M_TOK, D_MODEL, D_CONV);

    // 6) h = rmsnorm(resid, w_norm2)
    k_rmsnorm<<<M_TOK, 256, 0, stream>>>(resid, wn2, hbuf);

    // 7) act = silu(h @ w_gate^T) * (h @ w_up^T)
    k_gemm_wmma<2><<<dim3(FFN_INNER / 64, M_TOK / 128), 256, 0, stream>>>(
        hbuf, wgateT, wupT, nullptr, actb, M_TOK, FFN_INNER, D_MODEL);

    // 8) out = resid + act @ w_down^T
    k_gemm_wmma<1><<<dim3(D_MODEL / 128, M_TOK / 128), 256, 0, stream>>>(
        actb, wdownT, nullptr, resid, out, M_TOK, D_MODEL, FFN_INNER);
}
